// ConvolutionN_26517128085980
// MI455X (gfx1250) — matrix operand order repaired and checked by reading, not compiled, not run on hardware
//
#include <hip/hip_runtime.h>

typedef __attribute__((ext_vector_type(16))) _Float16 v16h;
typedef __attribute__((ext_vector_type(8)))  _Float16 v8h;
typedef __attribute__((ext_vector_type(8)))  float    v8f;
typedef __attribute__((ext_vector_type(4)))  float    v4f;

namespace {
constexpr int Bn = 64, Cn = 64, Hn = 128, Wn = 128;
constexpr int KK = 7, PD = 3, NTAP = KK * KK;   // 49 taps
constexpr int TOH = 16, TOW = 16;               // output tile per block
constexpr int HR = TOH + 2 * PD;                // 22 halo rows
constexpr int HC = TOW + 2 * PD;                // 22 halo cols
constexpr int XC = 32;                          // padded halo cols = 2 ow-blocks
constexpr int NBLK = XC / 16;                   // 2
constexpr int TPP = 16;                         // taps per pass (WMMA N)
constexpr int NPASS = 4;                        // 64 tap slots >= 49
constexpr int CH  = 4;                          // channels per async chunk
constexpr int NCH = Cn / CH;                    // 16 chunks
constexpr int CHUNK = CH * HR * XC;             // 2816 floats per chunk
constexpr int ITERS = CHUNK / 256;              // 11 async issues / thread / chunk
constexpr int SBP = HR * XC + 4;                // padded S row (tap stride): bank-spread
}

// Async global->LDS staging (raw fp32), tracked by ASYNCcnt.
#define ASYNC_LOAD_F32(gp, loff) \
  asm volatile("global_load_async_to_lds_b32 %0, %1, off" :: "v"(loff), "v"(gp) : "memory")
#define WAIT_ASYNCCNT(imm) asm volatile("s_wait_asynccnt " imm ::: "memory")

// x tile record = 128B (64 f16 channels); swizzle 16B granule inside record so
// A-loads (lanes m=0..15, granule slot s) and staging stores spread over banks.
__device__ __forceinline__ int xs_swz(int rec, int c) {
  return (rec << 6) + ((((c >> 3) ^ (rec & 7)) << 3) | (c & 7));
}

// S-matrix conv: S[pixel, tap] = sum_c x_f16[c,pixel] * w_f16[c,tap] via
// v_wmma_f32_16x16x32_f16 (M=16 ow, N=16 taps, K=32 channels, 2 per C=64),
// then out = bias + 49 shifted fp32 S reads from LDS.
__global__ __launch_bounds__(256)
void conv7x7_s_wmma(const float* __restrict__ xg,
                    const float* __restrict__ wg,
                    const float* __restrict__ bg,
                    float* __restrict__ og) {
  __shared__ _Float16 xs[HR * XC * Cn];     // 90112 B, swizzled records
  __shared__ float    sb[TPP][SBP];         // 45312 B, tap-stride padded
  __shared__ float    stage[2][CHUNK];      // 22528 B, async double buffer

  const int tid  = threadIdx.x;
  const int lane = tid & 31;
  const int wv   = __builtin_amdgcn_readfirstlane(tid >> 5);  // scalar wave id
  const int m    = lane & 15;   // A row (ow in block) == D column (tap)
  const int hi   = lane >> 4;   // half-wave select

  const int b   = blockIdx.z;
  const int oh0 = blockIdx.y * TOH;
  const int ow0 = blockIdx.x * TOW;

  // ---- stage x halo: async fp32 -> LDS (double buffered), convert to f16 ----
  auto issue = [&](int q) {
    const int buf = q & 1;
#pragma unroll
    for (int ii = 0; ii < ITERS; ++ii) {
      const int i   = tid + ii * 256;
      const int cl  = i / (HR * XC);
      const int rem = i - cl * (HR * XC);
      const int r   = rem >> 5;
      const int col = rem & 31;
      int hh = oh0 + r - PD;   hh = hh < 0 ? 0 : (hh > Hn - 1 ? Hn - 1 : hh);
      int ww = ow0 + col - PD; ww = ww < 0 ? 0 : (ww > Wn - 1 ? Wn - 1 : ww);
      const int c = q * CH + cl;
      const float* gp = &xg[((b * Cn + c) * Hn + hh) * Wn + ww];  // clamped, safe
      const unsigned loff = (unsigned)(uintptr_t)(const void*)&stage[buf][i];
      ASYNC_LOAD_F32(gp, loff);   // 11 per wave per chunk, uniform
    }
  };
  auto convert = [&](int q) {
    const int buf = q & 1;
#pragma unroll
    for (int ii = 0; ii < ITERS; ++ii) {
      const int i   = tid + ii * 256;
      const int cl  = i / (HR * XC);
      const int rem = i - cl * (HR * XC);
      const int r   = rem >> 5;
      const int col = rem & 31;
      const int hh = oh0 + r - PD;
      const int ww = ow0 + col - PD;
      const bool valid = (hh >= 0) && (hh < Hn) && (ww >= 0) && (ww < Wn);
      const float v = valid ? stage[buf][i] : 0.0f;   // zero-fill == PAD=3
      xs[xs_swz(r * XC + col, q * CH + cl)] = (_Float16)v;
    }
  };

  issue(0);
  for (int q = 0; q < NCH; ++q) {
    if (q + 1 < NCH) {
      issue(q + 1);
      WAIT_ASYNCCNT("11");   // only newest chunk may remain outstanding
    } else {
      WAIT_ASYNCCNT("0");
    }
    convert(q);
  }
  __syncthreads();

  float acc = 0.0f;
  const int owl = tid & 15;   // this thread's output col in tile
  const int ohl = tid >> 4;   // this thread's output row in tile

#pragma unroll
  for (int T = 0; T < NPASS; ++T) {
    // ---- B (filter) matrices for this 16-tap tile, held in VGPRs ----
    // B 32x16 f16 layout: N=lane%16; elements 0..7 -> K=8*hi+i, 8..15 -> K=16+8*hi+(i-8).
    const int tap = T * TPP + m;
    v16h bm0, bm1;
#pragma unroll
    for (int i = 0; i < 16; ++i) {
      const int k = (i & 7) + hi * 8 + (i >> 3) * 16;
      float w0 = 0.0f, w1 = 0.0f;
      if (tap < NTAP) {
        w0 = wg[k * NTAP + tap];          // c = k
        w1 = wg[(k + 32) * NTAP + tap];   // c = k + 32
      }
      bm0[i] = (_Float16)w0;
      bm1[i] = (_Float16)w1;
    }

    // ---- S[pixel, tap-tile] for all halo pixels (scalar loop -> EXEC=ones) ----
    for (int u = wv; u < HR * NBLK; u += 8) {
      const int r    = u >> 1;
      const int col0 = (u & 1) * 16;
      const int rec  = r * XC + col0 + m;
      const _Float16* base = &xs[rec << 6];
      const int sw = rec & 7;
      // A 16x32 f16: elements 0..7 <-> c granule (hi), 8..15 <-> granule (2+hi)
      v8h a0lo = *(const v8h*)(base + (((0 + hi) ^ sw) << 3));
      v8h a0hi = *(const v8h*)(base + (((2 + hi) ^ sw) << 3));
      v8h a1lo = *(const v8h*)(base + (((4 + hi) ^ sw) << 3));
      v8h a1hi = *(const v8h*)(base + (((6 + hi) ^ sw) << 3));
      v16h a0 = __builtin_shufflevector(a0lo, a0hi, 0,1,2,3,4,5,6,7,8,9,10,11,12,13,14,15);
      v16h a1 = __builtin_shufflevector(a1lo, a1hi, 0,1,2,3,4,5,6,7,8,9,10,11,12,13,14,15);
      v8f s = {};
      s = __builtin_amdgcn_wmma_f32_16x16x32_f16(false, a0, false, bm0, (short)0, s, false, false);
      s = __builtin_amdgcn_wmma_f32_16x16x32_f16(false, a1, false, bm1, (short)0, s, false, false);
      // D: lane n = m (tap), VGPR v -> pixel M = v + 8*hi
      float* sp = &sb[m][r * XC + col0 + hi * 8];
      *(v4f*)(sp)     = __builtin_shufflevector(s, s, 0, 1, 2, 3);
      *(v4f*)(sp + 4) = __builtin_shufflevector(s, s, 4, 5, 6, 7);
    }
    __syncthreads();

    // ---- accumulate the shifted taps of this pass ----
#pragma unroll
    for (int t = 0; t < TPP; ++t) {
      const int rt = T * TPP + t;
      if (rt < NTAP) {
        const int dh = rt / KK, dw = rt % KK;
        acc += sb[t][(ohl + dh) * XC + (owl + dw)];
      }
    }
    __syncthreads();   // before next pass overwrites sb
  }

  og[(b * Hn + oh0 + ohl) * Wn + ow0 + owl] = acc + bg[0];
}

extern "C" void kernel_launch(void* const* d_in, const int* in_sizes, int n_in,
                              void* d_out, int out_size, void* d_ws, size_t ws_size,
                              hipStream_t stream) {
  const float* x    = (const float*)d_in[0];   // (64,64,128,128) fp32
  const float* w    = (const float*)d_in[1];   // (64,7,7) fp32
  const float* bias = (const float*)d_in[2];   // (1,) fp32
  float* out = (float*)d_out;                  // (64,128,128) fp32

  dim3 grid(Wn / TOW, Hn / TOH, Bn);           // (8, 8, 64)
  conv7x7_s_wmma<<<grid, dim3(256, 1, 1), 0, stream>>>(x, w, bias, out);
}
